// ModernLorentzian_76871324664000
// MI455X (gfx1250) — compile-verified, matrix operand order checked by reading, not run on hardware
//
#include <hip/hip_runtime.h>
#include <hip/hip_bf16.h>

typedef __attribute__((ext_vector_type(16))) _Float16 v16h;
typedef __attribute__((ext_vector_type(8)))  float    v8f;

namespace {
constexpr int    T      = 131072;
constexpr int    THA    = 1024;
constexpr int    CH     = T / THA;     // 128 elements per thread in scan kernel
constexpr int    M      = 2000;
constexpr int    NCHUNK = M / 16;      // 125
constexpr int    HID    = 64;
constexpr int    OUTC   = 32;
constexpr int    NBLK   = T / 256;     // 512

constexpr size_t OT        = (size_t)T;
constexpr size_t OFF_H     = 0;
constexpr size_t OFF_L     = OT;
constexpr size_t OFF_C     = 2 * OT;
constexpr size_t OFF_B0    = 3 * OT;
constexpr size_t OFF_B1    = 4 * OT;
constexpr size_t OFF_B2    = 5 * OT;
constexpr size_t OFF_B3    = 6 * OT;
constexpr size_t OFF_F     = 7 * OT;          // 6 feature columns, column-major
constexpr size_t OFF_VOL   = 13 * OT;
constexpr size_t OFF_MASK  = 14 * OT;
constexpr size_t OFF_EMEAN = 15 * OT;
constexpr size_t OFF_SIG   = 16 * OT;
constexpr size_t OFF_SC    = 17 * OT;         // [0]=vol_sum; +8 scale1(64); +72 shift1(64); +136 scale2(32); +168 shift2(32)
constexpr size_t OFF_SCALE1 = OFF_SC + 8;
constexpr size_t OFF_SHIFT1 = OFF_SC + 72;
constexpr size_t OFF_SCALE2 = OFF_SC + 136;
constexpr size_t OFF_SHIFT2 = OFF_SC + 168;
constexpr size_t OFF_PS1   = OFF_SC + 256;            // 512*64
constexpr size_t OFF_PQ1   = OFF_PS1 + (size_t)NBLK * HID;
constexpr size_t OFF_PS2   = OFF_PQ1 + (size_t)NBLK * HID;  // 512*32
constexpr size_t OFF_PQ2   = OFF_PS2 + (size_t)NBLK * OUTC;
constexpr size_t OFF_MEMN  = OFF_PQ2 + (size_t)NBLK * OUTC; // 2000*32 f32
constexpr size_t OFF_MEMNB_F = OFF_MEMN + (size_t)M * OUTC; // 125*32*16 halves = 32000 f32
constexpr size_t OFF_EXTA_F  = OFF_MEMNB_F + 32000;         // T*32 halves = T*16 f32
} // namespace

// ---------------------------------------------------------------------------
// Scan helpers (single 1024-thread workgroup; T = 1024 * 128)
// ---------------------------------------------------------------------------
__device__ __forceinline__ void ema_pass(const float* x, float* y, float alpha,
                                         float2* sc, int tid) {
  const float om = 1.0f - alpha;
  const float x0 = x[0];                 // y[-1] := x[0] (matches lax.scan init)
  const int   i0 = tid * CH;
  float A = 1.0f, B = 0.0f;
  for (int j = 0; j < CH; ++j) { A *= om; B = alpha * x[i0 + j] + om * B; }
  sc[tid] = make_float2(A, B);
  __syncthreads();
  for (int off = 1; off < THA; off <<= 1) {
    float2 p = (tid >= off) ? sc[tid - off] : make_float2(1.0f, 0.0f);
    __syncthreads();
    float2 cur = sc[tid];
    sc[tid] = make_float2(cur.x * p.x, cur.x * p.y + cur.y);   // cur ∘ p
    __syncthreads();
  }
  float c;
  if (tid == 0) c = x0;
  else { float2 e = sc[tid - 1]; c = e.x * x0 + e.y; }
  for (int j = 0; j < CH; ++j) { c = alpha * x[i0 + j] + om * c; y[i0 + j] = c; }
  __syncthreads();
}

__device__ __forceinline__ void cumsum_pass(const float* x, float* y,
                                            float* sr, int tid) {
  const int i0 = tid * CH;
  float s = 0.0f;
  for (int j = 0; j < CH; ++j) s += x[i0 + j];
  sr[tid] = s;
  __syncthreads();
  for (int off = 1; off < THA; off <<= 1) {
    float p = (tid >= off) ? sr[tid - off] : 0.0f;
    __syncthreads();
    sr[tid] += p;
    __syncthreads();
  }
  float c = (tid == 0) ? 0.0f : sr[tid - 1];
  for (int j = 0; j < CH; ++j) { c += x[i0 + j]; y[i0 + j] = c; }
  __syncthreads();
}

// SMA with reference's front-pad-by-x0 semantics, from inclusive cumsum
__device__ __forceinline__ void sma_from_cs(const float* cs, float x0, int w,
                                            float* y, int tid) {
  const float inv = 1.0f / (float)w;
  for (int t = tid; t < T; t += THA) {
    float s = cs[t];
    if (t >= w)     s -= cs[t - w];
    if (t < w - 1)  s += (float)(w - 1 - t) * x0;
    y[t] = s * inv;
  }
  __syncthreads();
}

// ---------------------------------------------------------------------------
// Kernel A: technical indicators -> feats, vol, gate mask, vol_sum
// ---------------------------------------------------------------------------
__global__ __launch_bounds__(THA) void featureKernel(const float* __restrict__ x,
                                                     float* __restrict__ W) {
  __shared__ float2 sc[THA];
  float* red = (float*)sc;
  const int tid = threadIdx.x;

  float* H  = W + OFF_H;   float* L  = W + OFF_L;   float* C  = W + OFF_C;
  float* B0 = W + OFF_B0;  float* B1 = W + OFF_B1;  float* B2 = W + OFF_B2;
  float* B3 = W + OFF_B3;
  float* F0 = W + OFF_F;            float* F1 = W + OFF_F + OT;
  float* F2 = W + OFF_F + 2 * OT;   float* F3 = W + OFF_F + 3 * OT;
  float* F4 = W + OFF_F + 4 * OT;   float* F5 = W + OFF_F + 5 * OT;
  float* VOL = W + OFF_VOL;  float* MASK = W + OFF_MASK;

  // extract OHLC columns
  for (int t = tid; t < T; t += THA) {
    H[t] = x[t * 5 + 1]; L[t] = x[t * 5 + 2]; C[t] = x[t * 5 + 3];
  }
  __syncthreads();

  // ---- wavetrend ----
  for (int t = tid; t < T; t += THA) B0[t] = (H[t] + L[t] + C[t]) * (1.0f / 3.0f);
  __syncthreads();
  ema_pass(B0, B1, 2.0f / 11.0f, sc, tid);                       // esa
  for (int t = tid; t < T; t += THA) B2[t] = fabsf(B0[t] - B1[t]);
  __syncthreads();
  ema_pass(B2, B3, 2.0f / 11.0f, sc, tid);                       // d
  for (int t = tid; t < T; t += THA)
    B2[t] = (B0[t] - B1[t]) / (0.015f * B3[t] + 1e-10f);         // ci
  __syncthreads();
  ema_pass(B2, F1, 2.0f / 12.0f, sc, tid);                       // wt1
  cumsum_pass(F1, B0, red, tid);
  sma_from_cs(B0, F1[0], 4, F2, tid);                            // wt2

  // ---- rsi14 / rsi9 (ret = diff(close), d[0]=0) ----
  for (int t = tid; t < T; t += THA) B0[t] = (t > 0) ? (C[t] - C[t - 1]) : 0.0f;
  __syncthreads();
  for (int t = tid; t < T; t += THA) B1[t] = fmaxf(B0[t], 0.0f);
  __syncthreads();
  ema_pass(B1, B2, 1.0f / 14.0f, sc, tid);
  for (int t = tid; t < T; t += THA) B1[t] = fmaxf(-B0[t], 0.0f);
  __syncthreads();
  ema_pass(B1, B3, 1.0f / 14.0f, sc, tid);
  for (int t = tid; t < T; t += THA) {
    float rs = B2[t] / (B3[t] + 1e-10f);
    F0[t] = 100.0f - 100.0f / (1.0f + rs);
  }
  __syncthreads();
  for (int t = tid; t < T; t += THA) B1[t] = fmaxf(B0[t], 0.0f);
  __syncthreads();
  ema_pass(B1, B2, 1.0f / 9.0f, sc, tid);
  for (int t = tid; t < T; t += THA) B1[t] = fmaxf(-B0[t], 0.0f);
  __syncthreads();
  ema_pass(B1, B3, 1.0f / 9.0f, sc, tid);
  for (int t = tid; t < T; t += THA) {
    float rs = B2[t] / (B3[t] + 1e-10f);
    F5[t] = 100.0f - 100.0f / (1.0f + rs);
  }
  __syncthreads();

  // ---- vol = rolling_std(ret, 20) (ret still in B0) ----
  cumsum_pass(B0, B1, red, tid);
  sma_from_cs(B1, 0.0f, 20, B2, tid);                            // rm (ret[0]=0)
  for (int t = tid; t < T; t += THA) { float d = B0[t] - B2[t]; B3[t] = d * d; }
  __syncthreads();
  cumsum_pass(B3, B1, red, tid);
  sma_from_cs(B1, B3[0], 20, B2, tid);                           // var
  for (int t = tid; t < T; t += THA) VOL[t] = sqrtf(B2[t] + 1e-8f);
  __syncthreads();

  // ---- cci ----
  for (int t = tid; t < T; t += THA) B0[t] = (H[t] + L[t] + C[t]) * (1.0f / 3.0f);
  __syncthreads();
  cumsum_pass(B0, B1, red, tid);
  sma_from_cs(B1, B0[0], 20, B2, tid);                           // m
  for (int t = tid; t < T; t += THA) B3[t] = fabsf(B0[t] - B2[t]);
  __syncthreads();
  cumsum_pass(B3, B1, red, tid);
  sma_from_cs(B1, B3[0], 20, F3, tid);                           // md -> F3
  for (int t = tid; t < T; t += THA)
    F3[t] = (B0[t] - B2[t]) / (0.015f * F3[t] + 1e-10f);
  __syncthreads();

  // ---- adx ----
  for (int t = tid; t < T; t += THA) {
    B0[t] = (t > 0) ? (H[t] - H[t - 1]) : 0.0f;
    B1[t] = (t > 0) ? (L[t - 1] - L[t]) : 0.0f;
  }
  __syncthreads();
  for (int t = tid; t < T; t += THA) {
    float up = B0[t], dn = B1[t];
    B2[t] = (up > dn && up > 0.0f) ? up : 0.0f;                  // pdm
    B3[t] = (dn > up && dn > 0.0f) ? dn : 0.0f;                  // mdm
  }
  __syncthreads();
  for (int t = tid; t < T; t += THA) {
    float pc = (t > 0) ? C[t - 1] : C[0];
    B0[t] = fmaxf(H[t] - L[t], fmaxf(fabsf(H[t] - pc), fabsf(L[t] - pc)));
  }
  __syncthreads();
  ema_pass(B0, B0, 1.0f / 20.0f, sc, tid);                       // atr (in place)
  ema_pass(B2, B2, 1.0f / 20.0f, sc, tid);                       // ema(pdm)
  ema_pass(B3, B3, 1.0f / 20.0f, sc, tid);                       // ema(mdm)
  for (int t = tid; t < T; t += THA) {
    float pdi = 100.0f * B2[t] / (B0[t] + 1e-10f);
    float mdi = 100.0f * B3[t] / (B0[t] + 1e-10f);
    B1[t] = 100.0f * fabsf(pdi - mdi) / (pdi + mdi + 1e-10f);    // dx
  }
  __syncthreads();
  ema_pass(B1, F4, 1.0f / 20.0f, sc, tid);                       // adx

  // ---- close > sma(close,50) gate && adx>20 gate ----
  cumsum_pass(C, B0, red, tid);
  sma_from_cs(B0, C[0], 50, B1, tid);
  for (int t = tid; t < T; t += THA)
    MASK[t] = ((C[t] > B1[t]) && (F4[t] > 20.0f)) ? 1.0f : 0.0f;
  __syncthreads();

  // ---- deterministic vol sum ----
  float s = 0.0f;
  for (int t = tid; t < T; t += THA) s += VOL[t];
  red[tid] = s;
  __syncthreads();
  for (int off = THA / 2; off > 0; off >>= 1) {
    if (tid < off) red[tid] += red[tid + off];
    __syncthreads();
  }
  if (tid == 0) W[OFF_SC] = red[0];
}

// ---------------------------------------------------------------------------
// MLP helpers
// ---------------------------------------------------------------------------
__device__ __forceinline__ void mlp_h(const float* sW1, const float* sb1,
                                      const float f[6], float h[HID]) {
#pragma unroll
  for (int c = 0; c < HID; ++c) {
    float acc = sb1[c];
#pragma unroll
    for (int k = 0; k < 6; ++k) acc = fmaf(sW1[c * 6 + k], f[k], acc);
    h[c] = acc;
  }
}

// Kernel B1: BN1 batch statistics of h = feats @ W1^T + b1 (deterministic)
__global__ __launch_bounds__(256) void statsHKernel(const float* __restrict__ W1,
                                                    const float* __restrict__ b1,
                                                    float* __restrict__ W) {
  __shared__ float sW1[HID * 6];
  __shared__ float sb1[HID];
  __shared__ float part[2][HID * 8];
  const int tid = threadIdx.x;
  for (int i = tid; i < HID * 6; i += 256) sW1[i] = W1[i];
  if (tid < HID) sb1[tid] = b1[tid];
  __syncthreads();
  const int t = blockIdx.x * 256 + tid;
  float f[6];
#pragma unroll
  for (int k = 0; k < 6; ++k) f[k] = W[OFF_F + (size_t)k * OT + t];
  float h[HID];
  mlp_h(sW1, sb1, f, h);
  const int lane = tid & 31, wv = tid >> 5;
#pragma unroll
  for (int c = 0; c < HID; ++c) {
    float s = h[c], q = h[c] * h[c];
#pragma unroll
    for (int off = 16; off > 0; off >>= 1) {
      s += __shfl_down(s, off, 32);
      q += __shfl_down(q, off, 32);
    }
    if (lane == 0) { part[0][c * 8 + wv] = s; part[1][c * 8 + wv] = q; }
  }
  __syncthreads();
  if (tid < HID) {
    float s = 0.0f, q = 0.0f;
#pragma unroll
    for (int i = 0; i < 8; ++i) { s += part[0][tid * 8 + i]; q += part[1][tid * 8 + i]; }
    W[OFF_PS1 + (size_t)blockIdx.x * HID + tid] = s;
    W[OFF_PQ1 + (size_t)blockIdx.x * HID + tid] = q;
  }
}

__global__ void finalize1Kernel(const float* __restrict__ g1,
                                const float* __restrict__ be1,
                                float* __restrict__ W) {
  const int c = threadIdx.x;  // HID threads
  float s = 0.0f, q = 0.0f;
  for (int b = 0; b < NBLK; ++b) {
    s += W[OFF_PS1 + (size_t)b * HID + c];
    q += W[OFF_PQ1 + (size_t)b * HID + c];
  }
  const float mu  = s / (float)T;
  const float var = q / (float)T - mu * mu;
  const float sc  = g1[c] * rsqrtf(var + 1e-5f);
  W[OFF_SCALE1 + c] = sc;
  W[OFF_SHIFT1 + c] = be1[c] - mu * sc;
}

// Kernel B3: BN2 batch statistics of e = relu(bn1(h)) @ W2^T + b2
__global__ __launch_bounds__(256) void statsEKernel(const float* __restrict__ W1,
                                                    const float* __restrict__ b1,
                                                    const float* __restrict__ W2m,
                                                    const float* __restrict__ b2,
                                                    float* __restrict__ W) {
  __shared__ float sW1[HID * 6], sb1[HID], sc1[HID], sh1[HID];
  __shared__ float sW2[OUTC * HID], sb2[OUTC];
  __shared__ float part[2][OUTC * 8];
  const int tid = threadIdx.x;
  for (int i = tid; i < HID * 6; i += 256) sW1[i] = W1[i];
  for (int i = tid; i < OUTC * HID; i += 256) sW2[i] = W2m[i];
  if (tid < HID) { sb1[tid] = b1[tid]; sc1[tid] = W[OFF_SCALE1 + tid]; sh1[tid] = W[OFF_SHIFT1 + tid]; }
  if (tid < OUTC) sb2[tid] = b2[tid];
  __syncthreads();
  const int t = blockIdx.x * 256 + tid;
  float f[6];
#pragma unroll
  for (int k = 0; k < 6; ++k) f[k] = W[OFF_F + (size_t)k * OT + t];
  float h[HID];
  mlp_h(sW1, sb1, f, h);
#pragma unroll
  for (int c = 0; c < HID; ++c) h[c] = fmaxf(fmaf(sc1[c], h[c], sh1[c]), 0.0f);  // g
  float e[OUTC];
#pragma unroll
  for (int c = 0; c < OUTC; ++c) {
    float acc = sb2[c];
#pragma unroll
    for (int k = 0; k < HID; ++k) acc = fmaf(sW2[c * HID + k], h[k], acc);
    e[c] = acc;
  }
  const int lane = tid & 31, wv = tid >> 5;
#pragma unroll
  for (int c = 0; c < OUTC; ++c) {
    float s = e[c], q = e[c] * e[c];
#pragma unroll
    for (int off = 16; off > 0; off >>= 1) {
      s += __shfl_down(s, off, 32);
      q += __shfl_down(q, off, 32);
    }
    if (lane == 0) { part[0][c * 8 + wv] = s; part[1][c * 8 + wv] = q; }
  }
  __syncthreads();
  if (tid < OUTC) {
    float s = 0.0f, q = 0.0f;
#pragma unroll
    for (int i = 0; i < 8; ++i) { s += part[0][tid * 8 + i]; q += part[1][tid * 8 + i]; }
    W[OFF_PS2 + (size_t)blockIdx.x * OUTC + tid] = s;
    W[OFF_PQ2 + (size_t)blockIdx.x * OUTC + tid] = q;
  }
}

__global__ void finalize2Kernel(const float* __restrict__ g2,
                                const float* __restrict__ be2,
                                float* __restrict__ W) {
  const int c = threadIdx.x;  // OUTC threads
  float s = 0.0f, q = 0.0f;
  for (int b = 0; b < NBLK; ++b) {
    s += W[OFF_PS2 + (size_t)b * OUTC + c];
    q += W[OFF_PQ2 + (size_t)b * OUTC + c];
  }
  const float mu  = s / (float)T;
  const float var = q / (float)T - mu * mu;
  const float sc  = g2[c] * rsqrtf(var + 1e-5f);
  W[OFF_SCALE2 + c] = sc;
  W[OFF_SHIFT2 + c] = be2[c] - mu * sc;
}

// Kernel B5: ext = relu(bn2(e)); row mean; L2-normalize; store fp16 in WMMA A layout
__global__ __launch_bounds__(256) void extKernel(const float* __restrict__ W1,
                                                 const float* __restrict__ b1,
                                                 const float* __restrict__ W2m,
                                                 const float* __restrict__ b2,
                                                 float* __restrict__ W) {
  __shared__ float sW1[HID * 6], sb1[HID], sc1[HID], sh1[HID];
  __shared__ float sW2[OUTC * HID], sb2[OUTC], sc2[OUTC], sh2[OUTC];
  const int tid = threadIdx.x;
  for (int i = tid; i < HID * 6; i += 256) sW1[i] = W1[i];
  for (int i = tid; i < OUTC * HID; i += 256) sW2[i] = W2m[i];
  if (tid < HID) { sb1[tid] = b1[tid]; sc1[tid] = W[OFF_SCALE1 + tid]; sh1[tid] = W[OFF_SHIFT1 + tid]; }
  if (tid < OUTC) { sb2[tid] = b2[tid]; sc2[tid] = W[OFF_SCALE2 + tid]; sh2[tid] = W[OFF_SHIFT2 + tid]; }
  __syncthreads();
  const int t = blockIdx.x * 256 + tid;
  float f[6];
#pragma unroll
  for (int k = 0; k < 6; ++k) f[k] = W[OFF_F + (size_t)k * OT + t];
  float h[HID];
  mlp_h(sW1, sb1, f, h);
#pragma unroll
  for (int c = 0; c < HID; ++c) h[c] = fmaxf(fmaf(sc1[c], h[c], sh1[c]), 0.0f);
  float ext[OUTC];
  float mean = 0.0f, nrm2 = 0.0f;
#pragma unroll
  for (int c = 0; c < OUTC; ++c) {
    float acc = sb2[c];
#pragma unroll
    for (int k = 0; k < HID; ++k) acc = fmaf(sW2[c * HID + k], h[k], acc);
    float v = fmaxf(fmaf(sc2[c], acc, sh2[c]), 0.0f);
    ext[c] = v;
    mean += v;
    nrm2  = fmaf(v, v, nrm2);
  }
  mean *= (1.0f / (float)OUTC);
  W[OFF_EMEAN + t] = mean;
  const float inv = 1.0f / fmaxf(sqrtf(nrm2), 1e-12f);

  // WMMA 16-bit A-matrix lane layout: lane l (= row%16 [+16]) holds halves j:
  //   k = (j<8 ? j : j+8) + (l>=16 ? 8 : 0)
  _Float16* EXTA = (_Float16*)(W + OFF_EXTA_F);
  const int r = t >> 4, rowin = t & 15;
  const size_t base0 = ((size_t)r * 32 + rowin) * 16;
  const size_t base1 = ((size_t)r * 32 + rowin + 16) * 16;
#pragma unroll
  for (int j = 0; j < 16; ++j) {
    const int klo = (j < 8) ? j : j + 8;
    EXTA[base0 + j] = (_Float16)(ext[klo] * inv);
    EXTA[base1 + j] = (_Float16)(ext[klo + 8] * inv);
  }
}

// Normalize memory rows
__global__ void memNormKernel(const float* __restrict__ mem, float* __restrict__ W) {
  const int m = blockIdx.x * 256 + threadIdx.x;
  if (m >= M) return;
  float n = 0.0f;
#pragma unroll
  for (int k = 0; k < OUTC; ++k) { float v = mem[m * OUTC + k]; n = fmaf(v, v, n); }
  const float inv = 1.0f / fmaxf(sqrtf(n), 1e-12f);
#pragma unroll
  for (int k = 0; k < OUTC; ++k)
    W[OFF_MEMN + (size_t)m * OUTC + k] = mem[m * OUTC + k] * inv;
}

// Pre-swizzle mem_n into WMMA B-matrix lane layout, fp16, per 16-column chunk
__global__ void memPackKernel(float* __restrict__ W) {
  const int cidx = blockIdx.x;       // 0..124
  const int l    = threadIdx.x;      // 0..31
  _Float16* MB = (_Float16*)(W + OFF_MEMNB_F);
  const int col = cidx * 16 + (l & 15);
  const int kb  = (l >= 16) ? 8 : 0;
  const size_t base = ((size_t)cidx * 32 + l) * 16;
#pragma unroll
  for (int j = 0; j < 16; ++j) {
    const int k = ((j < 8) ? j : j + 8) + kb;
    MB[base + j] = (_Float16)W[OFF_MEMN + (size_t)col * OUTC + k];
  }
}

// ---------------------------------------------------------------------------
// Kernel C: fused similarity GEMM (WMMA fp16) + per-row top-8 + kNN signal.
// Single-wave workgroups: barrier ops are free (ISA 3.1), each wave owns a
// private 2 KB LDS staging tile for the C-matrix lane->row transpose.
// ---------------------------------------------------------------------------
__global__ __launch_bounds__(32) void simTopkKernel(const int* __restrict__ labels,
                                                    float* __restrict__ W) {
  __shared__ float sim[512];         // 2 tiles x 16x16 f32
  const int lane = threadIdx.x;
  const _Float16* EXTA = (const _Float16*)(W + OFF_EXTA_F);
  const _Float16* MB   = (const _Float16*)(W + OFF_MEMNB_F);

  const int tile0 = blockIdx.x * 2;  // this wave owns 32 rows (2 A tiles)
  const v16h a0 = *(const v16h*)(EXTA + ((size_t)tile0 * 32 + lane) * 16);
  const v16h a1 = *(const v16h*)(EXTA + ((size_t)(tile0 + 1) * 32 + lane) * 16);

  float bv[8];
  int   bi[8];
#pragma unroll
  for (int i = 0; i < 8; ++i) { bv[i] = -1e30f; bi[i] = 0; }

  const int col   = lane & 15;
  const int rbase = (lane >> 4) * 8;                       // C layout: M = r + 8*(lane>=16)
  // lane owns one full output row: tile (lane>>4), row-in-tile (lane&15)
  const float4* srow4 = (const float4*)(sim + (lane >> 4) * 256 + (lane & 15) * 16);

  for (int cidx = 0; cidx < NCHUNK; ++cidx) {
    const v16h b = *(const v16h*)(MB + ((size_t)cidx * 32 + lane) * 16);
    // prefetch next B chunk into cache (global_prefetch_b8)
    const int nxt = (cidx + 1 < NCHUNK) ? (cidx + 1) : cidx;
    __builtin_prefetch(MB + ((size_t)nxt * 32 + lane) * 16, 0, 1);

    v8f z = {};
    v8f c0 = __builtin_amdgcn_wmma_f32_16x16x32_f16(false, a0, false, b,
                                                    (short)0, z, false, false);
    v8f c1 = __builtin_amdgcn_wmma_f32_16x16x32_f16(false, a1, false, b,
                                                    (short)0, z, false, false);
#pragma unroll
    for (int r = 0; r < 8; ++r) {
      sim[(r + rbase) * 16 + col]       = c0[r];
      sim[256 + (r + rbase) * 16 + col] = c1[r];
    }
    __syncthreads();   // single-wave workgroup: S_NOP-class barrier

    // vectorized row fetch: 4x ds_load_b128 into registers
    const float4 q0 = srow4[0];
    const float4 q1 = srow4[1];
    const float4 q2 = srow4[2];
    const float4 q3 = srow4[3];
    const float vals[16] = { q0.x, q0.y, q0.z, q0.w,
                             q1.x, q1.y, q1.z, q1.w,
                             q2.x, q2.y, q2.z, q2.w,
                             q3.x, q3.y, q3.z, q3.w };
    const int nb = cidx * 16;
#pragma unroll
    for (int q = 0; q < 16; ++q) {
      const float v = vals[q];
      if (v > bv[7]) {
        bv[7] = v; bi[7] = nb + q;
#pragma unroll
        for (int p = 7; p > 0; --p) {
          if (bv[p] > bv[p - 1]) {
            float tv = bv[p]; bv[p] = bv[p - 1]; bv[p - 1] = tv;
            int   ti = bi[p]; bi[p] = bi[p - 1]; bi[p - 1] = ti;
          }
        }
      }
    }
    __syncthreads();
  }

  // negd = -dist = sim - 1 ; w = exp(negd * RW), RW = 8
  float wsum = 0.0f, lsum = 0.0f;
#pragma unroll
  for (int i = 0; i < 8; ++i) {
    const float w8 = __expf((bv[i] - 1.0f) * 8.0f);
    const float lf = (labels[bi[i]] > 0) ? 1.0f : -1.0f;
    wsum += w8;
    lsum = fmaf(lf, w8, lsum);
  }
  const float knn = lsum / (wsum + 1e-8f);
  const int t = blockIdx.x * 32 + lane;
  W[OFF_SIG + t] = 0.6f * W[OFF_EMEAN + t] + 0.4f * knn;
}

// ---------------------------------------------------------------------------
// Kernel D: sma(sig,2) + vol/trend/adx gates
// ---------------------------------------------------------------------------
__global__ void finalKernel(float* __restrict__ out, const float* __restrict__ W) {
  const int t = blockIdx.x * 256 + threadIdx.x;
  const float volmean = W[OFF_SC] / (float)T;
  const float sp = W[OFF_SIG + t];
  const float sm = (t > 0) ? W[OFF_SIG + t - 1] : sp;
  float sig = 0.5f * (sp + sm);
  sig *= W[OFF_MASK + t];
  sig *= (W[OFF_VOL + t] > volmean) ? 1.0f : 0.0f;
  out[t] = sig;
}

// ---------------------------------------------------------------------------
extern "C" void kernel_launch(void* const* d_in, const int* in_sizes, int n_in,
                              void* d_out, int out_size, void* d_ws, size_t ws_size,
                              hipStream_t stream) {
  (void)in_sizes; (void)n_in; (void)out_size; (void)ws_size;
  const float* x    = (const float*)d_in[0];
  const float* mem  = (const float*)d_in[1];
  const int*   lab  = (const int*)d_in[2];
  const float* W1   = (const float*)d_in[3];
  const float* b1   = (const float*)d_in[4];
  const float* g1   = (const float*)d_in[5];
  const float* be1  = (const float*)d_in[6];
  const float* W2m  = (const float*)d_in[7];
  const float* b2   = (const float*)d_in[8];
  const float* g2   = (const float*)d_in[9];
  const float* be2  = (const float*)d_in[10];
  float* W   = (float*)d_ws;
  float* out = (float*)d_out;

  featureKernel<<<1, THA, 0, stream>>>(x, W);
  memNormKernel<<<(M + 255) / 256, 256, 0, stream>>>(mem, W);
  memPackKernel<<<NCHUNK, 32, 0, stream>>>(W);
  statsHKernel<<<NBLK, 256, 0, stream>>>(W1, b1, W);
  finalize1Kernel<<<1, HID, 0, stream>>>(g1, be1, W);
  statsEKernel<<<NBLK, 256, 0, stream>>>(W1, b1, W2m, b2, W);
  finalize2Kernel<<<1, OUTC, 0, stream>>>(g2, be2, W);
  extKernel<<<NBLK, 256, 0, stream>>>(W1, b1, W2m, b2, W);
  simTopkKernel<<<T / 32, 32, 0, stream>>>(lab, W);
  finalKernel<<<NBLK, 256, 0, stream>>>(out, W);
}
